// LSTMModel3_70677981823289
// MI455X (gfx1250) — compile-verified, matrix operand order checked by reading, not run on hardware
//
#include <hip/hip_runtime.h>
#include <hip/hip_bf16.h>

#define DEVINL __device__ __forceinline__

typedef __attribute__((ext_vector_type(16))) __bf16 v16bf;
typedef __attribute__((ext_vector_type(8)))  float  v8f;

// ---------------- geometry ----------------
#define NB    256
#define LSEQ  128
#define INF   8
#define KT_H  24          // 768 / 32
#define KT_L1 25          // 768/32 + 1 concatenated input K-tile
#define ROWS  16
#define SHH   808         // LDS activation row stride in halves (>=800, 16B aligned, 20-bank skew)

// packed bf16 weight-fragment offsets (in halves) inside d_ws
#define OFF_G1  0u                              // layer1 combined: [Whh1 kt0..23 | Wih1 kt24] per n-tile
#define SZ_G1   (192u * 25u * 512u)
#define OFF_IH2 (OFF_G1 + SZ_G1)
#define SZ_HH   (192u * 24u * 512u)
#define OFF_HH2 (OFF_IH2 + SZ_HH)
#define OFF_W1  (OFF_HH2 + SZ_HH)
#define SZ_W1   (32u * 24u * 512u)
#define OFF_BC1 (OFF_W1 + SZ_W1)                // 3072 fp32 combined biases layer1 (in halves: 6144)
#define OFF_BC2 (OFF_BC1 + 6144u)               // 3072 fp32 combined biases layer2

#define WMMA_BF16(a, b, c) \
  __builtin_amdgcn_wmma_f32_16x16x32_bf16(false, (a), false, (b), (short)0, (c), false, false)

DEVINL unsigned short f2bf(float f) {
  unsigned u = __float_as_uint(f);
  unsigned r = (u + 0x7FFFu + ((u >> 16) & 1u)) >> 16;   // RNE
  return (unsigned short)r;
}

union Frag { unsigned short s[16]; uint4 u[2]; v16bf v; };

DEVINL v8f vzero() {
  v8f v;
#pragma unroll
  for (int i = 0; i < 8; ++i) v[i] = 0.f;
  return v;
}

DEVINL float sigm_f(float x) { return __builtin_amdgcn_rcpf(1.f + __expf(-x)); }
DEVINL float tanh_f(float x) { return 1.f - 2.f * __builtin_amdgcn_rcpf(1.f + __expf(2.f * x)); }

// A fragment (16x32 bf16, M rows across lanes) from LDS row-major activation buffer.
DEVINL v16bf load_afrag(const unsigned short* base, int kt, int lane) {
  int off = (lane & 15) * SHH + kt * 32 + ((lane >> 4) << 3);
  Frag f;
  f.u[0] = *(const uint4*)(base + off);        // K 0..7   (lanes>=16: K 8..15)
  f.u[1] = *(const uint4*)(base + off + 16);   // K 16..23 (lanes>=16: K 24..31)
  return f.v;
}

// B fragment (pre-swizzled 1024B blob): lane's 32 bytes are contiguous.
DEVINL v16bf load_bfrag(const unsigned short* w, unsigned frag, int lane) {
  const uint4* p = (const uint4*)(w + (size_t)frag * 512 + lane * 16);
  Frag f;
  f.u[0] = p[0];
  f.u[1] = p[1];
  return f.v;
}

// ---- one layer-1 gate tile: uniform K sweep over [h1 | inp] (25 K-tiles, 4 gates) ----
DEVINL void l1_tile(int jt, const unsigned short* Pg1,
                    const unsigned short* h1s, int lane, v8f (&acc)[4]) {
#pragma unroll
  for (int gg = 0; gg < 4; ++gg) acc[gg] = vzero();
#pragma unroll 1
  for (int kt = 0; kt < KT_L1; ++kt) {
    v16bf a = load_afrag(h1s, kt, lane);
#pragma unroll
    for (int gg = 0; gg < 4; ++gg) {
      v16bf b = load_bfrag(Pg1, (unsigned)((gg * 48 + jt) * KT_L1 + kt), lane);
      acc[gg] = WMMA_BF16(a, b, acc[gg]);
    }
  }
}

// ---- one layer-2 gate tile: acc = h1 x Wih2 + h2 x Whh2 (4 gates) ----
DEVINL void l2_tile(int jt,
                    const unsigned short* Pih2, const unsigned short* Phh2,
                    const unsigned short* h1s, const unsigned short* h2s,
                    int lane, v8f (&acc)[4]) {
#pragma unroll
  for (int gg = 0; gg < 4; ++gg) acc[gg] = vzero();
#pragma unroll 1
  for (int kt = 0; kt < KT_H; ++kt) {          // pass 1: h1(t) @ W_ih2^T
    v16bf a = load_afrag(h1s, kt, lane);
#pragma unroll
    for (int gg = 0; gg < 4; ++gg) {
      v16bf b = load_bfrag(Pih2, (unsigned)((gg * 48 + jt) * KT_H + kt), lane);
      acc[gg] = WMMA_BF16(a, b, acc[gg]);
    }
  }
#pragma unroll 1
  for (int kt = 0; kt < KT_H; ++kt) {          // pass 2: h2(t-1) @ W_hh2^T
    v16bf a = load_afrag(h2s, kt, lane);
#pragma unroll
    for (int gg = 0; gg < 4; ++gg) {
      v16bf b = load_bfrag(Phh2, (unsigned)((gg * 48 + jt) * KT_H + kt), lane);
      acc[gg] = WMMA_BF16(a, b, acc[gg]);
    }
  }
}

// LSTM elementwise for one 16(row)x16(cell) tile: updates c, produces h values
// in registers (LDS write deferred past the phase barrier by the caller).
DEVINL void lstm_elem(v8f (&acc)[4], float (&c)[8], float (&hv)[8],
                      const float* bc, int jt, int lane) {
  int jc = jt * 16 + (lane & 15);
  float bI = bc[jc];
  float bF = bc[768 + jc];
  float bG = bc[1536 + jc];
  float bO = bc[2304 + jc];
#pragma unroll
  for (int r = 0; r < 8; ++r) {
    float ig = sigm_f(acc[0][r] + bI);
    float fg = sigm_f(acc[1][r] + bF);
    float gg = tanh_f(acc[2][r] + bG);
    float og = sigm_f(acc[3][r] + bO);
    float cn = fg * c[r] + ig * gg;
    c[r] = cn;
    hv[r] = og * tanh_f(cn);
  }
}

// ---- pack fp32 (R x C, row = output feature) into WMMA B fragments ----
// dst fragment index = nt * KTO + kto + kt  (allows interleaving Whh1 + Wih1)
__global__ __launch_bounds__(256) void pack_b(const float* __restrict__ W,
                                              unsigned short* __restrict__ dst,
                                              int R, int C, int KT, int KTO, int kto) {
  int tid = blockIdx.x * blockDim.x + threadIdx.x;
  int total = (R >> 4) * KT * 512;
  if (tid >= total) return;
  int j  = tid & 15;
  int l  = (tid >> 4) & 31;
  int f  = tid >> 9;
  int kt = f % KT;
  int nt = f / KT;
  int n  = nt * 16 + (l & 15);
  int chunk = j >> 3, idx = j & 7;
  int k = kt * 32 + ((l < 16) ? (chunk ? 16 + idx : idx)
                              : (chunk ? 24 + idx : 8 + idx));
  float v = (k < C) ? W[(size_t)n * C + k] : 0.f;
  dst[((size_t)nt * KTO + kto + kt) * 512 + l * 16 + j] = f2bf(v);
}

// ---- combine gate biases: bc = b_ih + b_hh (fp32) ----
__global__ __launch_bounds__(256) void bias_comb(const float* __restrict__ a,
                                                 const float* __restrict__ b,
                                                 float* __restrict__ dst, int n) {
  int tid = blockIdx.x * blockDim.x + threadIdx.x;
  if (tid < n) dst[tid] = a[tid] + b[tid];
}

// ---------------- main persistent LSTM kernel: 16 blocks x 768 threads ----------------
__global__ __launch_bounds__(768, 1) void lstm_main(
    const float* __restrict__ x,
    const float* __restrict__ b1, const float* __restrict__ W2,
    const float* __restrict__ b2,
    const unsigned short* __restrict__ wp,
    float* __restrict__ out) {
  __shared__ unsigned short h1s[ROWS * SHH];   // cols 0..767 = h1, 768..799 = [x_t, store, 0-pad]
  __shared__ unsigned short h2s[ROWS * SHH];
  __shared__ float out_acc[ROWS], out_prev[ROWS], store_prev[ROWS];

  const int g    = blockIdx.x;       // batch group: rows g*16 .. g*16+15
  const int tid  = threadIdx.x;
  const int lane = tid & 31;
  const int wid  = tid >> 5;         // 0..23

  for (int i = tid; i < ROWS * SHH; i += blockDim.x) { h1s[i] = 0; h2s[i] = 0; }
  if (tid < ROWS) { out_prev[tid] = 0.f; store_prev[tid] = 0.f; }

  float c1[2][8], c2[2][8];
#pragma unroll
  for (int jj = 0; jj < 2; ++jj)
#pragma unroll
    for (int r = 0; r < 8; ++r) { c1[jj][r] = 0.f; c2[jj][r] = 0.f; }
  __syncthreads();

  const unsigned short* Pg1  = wp + OFF_G1;
  const unsigned short* Pih2 = wp + OFF_IH2;
  const unsigned short* Phh2 = wp + OFF_HH2;
  const unsigned short* Pw1  = wp + OFF_W1;
  const float* bc1 = (const float*)(wp + OFF_BC1);
  const float* bc2 = (const float*)(wp + OFF_BC2);

  const int jcb = (lane & 15);          // column within tile
  const int mb  = (lane >> 4) << 3;     // row base for D-layout
#pragma unroll 1
  for (int t = 0; t < LSEQ; ++t) {
    // ---- step prep: write x_t into A columns 768..775, feedback store into col 776 ----
    if (tid < 128) {
      int m = tid >> 3, k = tid & 7;
      float xv = x[((size_t)(g * ROWS + m) * LSEQ + t) * INF + k];
      h1s[m * SHH + 768 + k] = f2bf(xv);
    }
    if (tid < ROWS) {
      int m = tid;
      float x0 = x[((size_t)(g * ROWS + m) * LSEQ + t) * INF];
      float s = store_prev[m] + x0 - out_prev[m];
      store_prev[m] = s;
      h1s[m * SHH + 776] = f2bf(s);
      out[(size_t)NB * LSEQ + (size_t)(g * ROWS + m) * LSEQ + t] = s;  // stores output
      out_acc[m] = 0.f;
    }
    __syncthreads();

    // ================= layer 1 (two j-tiles, manually unrolled) =================
    float h1v[2][8];
    {
      v8f acc[4];
      l1_tile(wid,      Pg1, h1s, lane, acc);
      lstm_elem(acc, c1[0], h1v[0], bc1, wid,      lane);
      l1_tile(wid + 24, Pg1, h1s, lane, acc);
      lstm_elem(acc, c1[1], h1v[1], bc1, wid + 24, lane);
    }
    __syncthreads();                               // all reads of h1(t-1) complete
#pragma unroll
    for (int jj = 0; jj < 2; ++jj) {
      int jc = (wid + jj * 24) * 16 + jcb;
#pragma unroll
      for (int r = 0; r < 8; ++r) h1s[(mb + r) * SHH + jc] = f2bf(h1v[jj][r]);
    }
    __syncthreads();                               // h1(t) published

    // ================= layer 2 (two j-tiles, manually unrolled) =================
    float h2v[2][8];
    {
      v8f acc[4];
      l2_tile(wid,      Pih2, Phh2, h1s, h2s, lane, acc);
      lstm_elem(acc, c2[0], h2v[0], bc2, wid,      lane);
      l2_tile(wid + 24, Pih2, Phh2, h1s, h2s, lane, acc);
      lstm_elem(acc, c2[1], h2v[1], bc2, wid + 24, lane);
    }
    __syncthreads();                               // all reads of h2(t-1) complete
#pragma unroll
    for (int jj = 0; jj < 2; ++jj) {
      int jc = (wid + jj * 24) * 16 + jcb;
#pragma unroll
      for (int r = 0; r < 8; ++r) h2s[(mb + r) * SHH + jc] = f2bf(h2v[jj][r]);
    }
    __syncthreads();                               // h2(t) published

    // ================= MLP head =================
#pragma unroll 1
    for (int nt = wid; nt < 32; nt += 24) {
      v8f am = vzero();
#pragma unroll 1
      for (int kt = 0; kt < KT_H; ++kt) {
        v16bf a = load_afrag(h2s, kt, lane);
        v16bf b = load_bfrag(Pw1, (unsigned)(nt * KT_H + kt), lane);
        am = WMMA_BF16(a, b, am);
      }
      int n = nt * 16 + jcb;
      float bb = b1[n];
      float wv = W2[n];
#pragma unroll
      for (int r = 0; r < 8; ++r) {
        float z = am[r] + bb;
        z = fmaxf(z, 0.f);
        atomicAdd(&out_acc[mb + r], z * wv);       // ds_add_f32
      }
    }
    __syncthreads();

    if (tid < ROWS) {
      float o = out_acc[tid] + b2[0];
      out[(size_t)(g * ROWS + tid) * LSEQ + t] = o;  // outs output
      out_prev[tid] = o;
    }
    __syncthreads();
  }
}

extern "C" void kernel_launch(void* const* d_in, const int* in_sizes, int n_in,
                              void* d_out, int out_size, void* d_ws, size_t ws_size,
                              hipStream_t stream) {
  (void)in_sizes; (void)n_in; (void)out_size; (void)ws_size;
  const float* x    = (const float*)d_in[0];
  const float* Wih1 = (const float*)d_in[1];
  const float* bih1 = (const float*)d_in[2];
  const float* Whh1 = (const float*)d_in[3];
  const float* bhh1 = (const float*)d_in[4];
  const float* Wih2 = (const float*)d_in[5];
  const float* bih2 = (const float*)d_in[6];
  const float* Whh2 = (const float*)d_in[7];
  const float* bhh2 = (const float*)d_in[8];
  const float* W1   = (const float*)d_in[9];
  const float* b1   = (const float*)d_in[10];
  const float* W2   = (const float*)d_in[11];
  const float* b2   = (const float*)d_in[12];
  unsigned short* ws = (unsigned short*)d_ws;
  float* out = (float*)d_out;

  // pack all weights into bf16 WMMA B fragments (deterministic, re-done every call)
  {
    int total = 192 * 24 * 512;      // Whh1 -> combined layer1 buffer, kt 0..23
    pack_b<<<(total + 255) / 256, 256, 0, stream>>>(Whh1, ws + OFF_G1, 3072, 768, 24, 25, 0);
  }
  {
    int total = 192 * 1 * 512;       // Wih1 (K=9, padded) -> combined layer1 buffer, kt 24
    pack_b<<<(total + 255) / 256, 256, 0, stream>>>(Wih1, ws + OFF_G1, 3072, 9, 1, 25, 24);
  }
  {
    int total = 192 * 24 * 512;
    pack_b<<<(total + 255) / 256, 256, 0, stream>>>(Wih2, ws + OFF_IH2, 3072, 768, 24, 24, 0);
    pack_b<<<(total + 255) / 256, 256, 0, stream>>>(Whh2, ws + OFF_HH2, 3072, 768, 24, 24, 0);
  }
  {
    int total = 32 * 24 * 512;
    pack_b<<<(total + 255) / 256, 256, 0, stream>>>(W1, ws + OFF_W1, 512, 768, 24, 24, 0);
  }
  bias_comb<<<12, 256, 0, stream>>>(bih1, bhh1, (float*)(ws + OFF_BC1), 3072);
  bias_comb<<<12, 256, 0, stream>>>(bih2, bhh2, (float*)(ws + OFF_BC2), 3072);

  lstm_main<<<16, 768, 0, stream>>>(x, b1, W2, b2, ws, out);
}
